// BaselineSpanScorer_19301583028867
// MI455X (gfx1250) — matrix operand order repaired and checked by reading, not compiled, not run on hardware
//
#include <hip/hip_runtime.h>

typedef __attribute__((ext_vector_type(16))) _Float16 v16h;
typedef __attribute__((ext_vector_type(8)))  _Float16 v8h;
typedef __attribute__((ext_vector_type(4)))  _Float16 v4h;
typedef __attribute__((ext_vector_type(8)))  float    v8f;

#define Bsz   32
#define Tlen  512
#define Dd    1024
#define TWO_D 2048
#define Hh    150
#define HP    160       // H padded to multiple of 16
#define LBL   17
#define LP    32        // labels padded to 2 x 16
#define MT    32        // spans per workgroup (2 x 16-row WMMA tiles)
#define KSTEPS (TWO_D / 32)   // 64 K-steps for GEMM1
#define NT1   (HP / 16)       // 10 hidden-column tiles -> 10 waves
#define KT2   (HP / 32)       // 5 K-steps for GEMM2
#define A_STRIDE (TWO_D + 8)  // LDS row stride (halves), +8 kills ds bank conflicts
#define H_STRIDE (HP + 8)

#define W1H_ENTRIES (KSTEPS * NT1 * 32)   // 20480 lane-entries of 16 halves
#define W2H_ENTRIES (KT2 * 2 * 32)        // 320

__device__ inline v8f wmma16x16x32(v16h a, v16h b, v8f c) {
  return __builtin_amdgcn_wmma_f32_16x16x32_f16(
      /*neg_a=*/false, a, /*neg_b=*/false, b,
      /*c_mod=*/(short)0, c, /*reuse_a=*/false, /*reuse_b=*/false);
}

// ---------------------------------------------------------------------------
// Prep: convert W1 [2048,150] f32 and W2 [150,17] f32 into zero-padded f16,
// pre-swizzled into the WMMA B-fragment layout:
//   entry e = (kt*numNT + nt)*32 + lane ; lane holds 16 contiguous halves:
//   halves[q] = W[k = kt*32 + (lane>>4)*8 + (q&7) + (q>>3)*16][n = nt*16 + (lane&15)]
// ---------------------------------------------------------------------------
__global__ void prep_weights(const float* __restrict__ W1,
                             const float* __restrict__ W2,
                             _Float16* __restrict__ W1h,
                             _Float16* __restrict__ W2h) {
  int e = blockIdx.x * blockDim.x + threadIdx.x;
  if (e < W1H_ENTRIES) {
    int lane = e & 31, tile = e >> 5;
    int nt = tile % NT1, kt = tile / NT1;
    int n = nt * 16 + (lane & 15);
    int kbase = kt * 32 + (lane >> 4) * 8;
    v16h v;
#pragma unroll
    for (int q = 0; q < 16; ++q) {
      int k = kbase + (q & 7) + (q >> 3) * 16;
      float f = (n < Hh) ? W1[k * Hh + n] : 0.0f;
      v[q] = (_Float16)f;
    }
    ((v16h*)W1h)[e] = v;
  } else if (e < W1H_ENTRIES + W2H_ENTRIES) {
    int e2 = e - W1H_ENTRIES;
    int lane = e2 & 31, tile = e2 >> 5;
    int nt = tile % 2, kt = tile / 2;
    int n = nt * 16 + (lane & 15);
    int kbase = kt * 32 + (lane >> 4) * 8;
    v16h v;
#pragma unroll
    for (int q = 0; q < 16; ++q) {
      int k = kbase + (q & 7) + (q >> 3) * 16;
      float f = (k < Hh && n < LBL) ? W2[k * LBL + n] : 0.0f;
      v[q] = (_Float16)f;
    }
    ((v16h*)W2h)[e2] = v;
  }
}

// ---------------------------------------------------------------------------
// Main: per workgroup, 32 spans. Gather+convert features to LDS (f16),
// GEMM1 via WMMA (10 waves x 16 hidden cols each), ReLU, GEMM2 via WMMA.
// ---------------------------------------------------------------------------
__global__ __launch_bounds__(320) void
span_mlp(const float* __restrict__ xs,
         const int*   __restrict__ spans,
         const int*   __restrict__ batch_ids,
         const _Float16* __restrict__ W1h,
         const float* __restrict__ b1,
         const _Float16* __restrict__ W2h,
         const float* __restrict__ b2,
         float* __restrict__ out,
         int N) {
  __shared__ _Float16 Ash[MT * A_STRIDE];   // 32 x 2056 halves = 128.5 KB
  __shared__ _Float16 Hsh[MT * H_STRIDE];   // 32 x 168 halves  = 10.5 KB
  __shared__ int flatIdx[MT * 2];

  const int tid    = threadIdx.x;
  const int block0 = blockIdx.x * MT;

  // --- flat gather indices (span + batch*T) ---
  if (tid < MT * 2) {
    int m = tid >> 1, part = tid & 1;
    int n = block0 + m;
    int idx = 0;
    if (n < N) idx = batch_ids[n] * Tlen + spans[2 * n + part];
    flatIdx[tid] = idx;
  }
  __syncthreads();

  // --- gather: 32 spans x 2 tokens x 256 float4 -> f16 rows in LDS ---
  const float4* xs4 = (const float4*)xs;
  for (int j = tid; j < MT * 2 * (Dd / 4); j += 320) {
    int m    = j >> 9;          // / 512
    int r    = j & 511;
    int part = r >> 8;
    int c    = r & 255;
    long flat = (long)flatIdx[m * 2 + part];
    float4 v = xs4[flat * (Dd / 4) + c];
    v4h h;
    h[0] = (_Float16)v.x; h[1] = (_Float16)v.y;
    h[2] = (_Float16)v.z; h[3] = (_Float16)v.w;
    *(v4h*)&Ash[m * A_STRIDE + part * Dd + c * 4] = h;
  }
  __syncthreads();

  const int wave   = tid >> 5;     // 0..9 == hidden n-tile for GEMM1
  const int lane   = tid & 31;
  const int lane15 = lane & 15;
  const int laneh  = lane >> 4;

  // --- GEMM1: [32 x 2048] x [2048 x 160], wave handles 16 cols, 2 M-tiles ---
  v8f acc0 = {}; v8f acc1 = {};
  const v16h* Bp = (const v16h*)W1h + wave * 32 + lane;
#pragma unroll 4
  for (int kt = 0; kt < KSTEPS; ++kt) {
    v16h b = Bp[kt * (NT1 * 32)];
    union { v16h v; v8h h[2]; } a0, a1;
    const _Float16* ar0 = &Ash[lane15 * A_STRIDE + kt * 32 + laneh * 8];
    a0.h[0] = *(const v8h*)ar0;
    a0.h[1] = *(const v8h*)(ar0 + 16);
    const _Float16* ar1 = ar0 + 16 * A_STRIDE;
    a1.h[0] = *(const v8h*)ar1;
    a1.h[1] = *(const v8h*)(ar1 + 16);
    acc0 = wmma16x16x32(a0.v, b, acc0);
    acc1 = wmma16x16x32(a1.v, b, acc1);
  }

  // --- bias + ReLU, store hidden tile to LDS as f16 (A-fragment friendly) ---
  {
    int ncol = wave * 16 + lane15;
    float bias = (ncol < Hh) ? b1[ncol] : 0.0f;
#pragma unroll
    for (int i = 0; i < 8; ++i) {
      int m = laneh * 8 + i;
      float x0 = acc0[i] + bias; x0 = x0 > 0.0f ? x0 : 0.0f;
      float x1 = acc1[i] + bias; x1 = x1 > 0.0f ? x1 : 0.0f;
      Hsh[m * H_STRIDE + ncol]        = (_Float16)x0;
      Hsh[(16 + m) * H_STRIDE + ncol] = (_Float16)x1;
    }
  }
  __syncthreads();

  // --- GEMM2: [32 x 160] x [160 x 32]; waves 0..3 = (mtile, ntile) pairs ---
  if (wave < 4) {
    const int mt  = wave >> 1;
    const int nt2 = wave & 1;
    v8f acc = {};
#pragma unroll
    for (int kt = 0; kt < KT2; ++kt) {
      v16h b = ((const v16h*)W2h)[(kt * 2 + nt2) * 32 + lane];
      union { v16h v; v8h h[2]; } a;
      const _Float16* ar = &Hsh[(mt * 16 + lane15) * H_STRIDE + kt * 32 + laneh * 8];
      a.h[0] = *(const v8h*)ar;
      a.h[1] = *(const v8h*)(ar + 16);
      acc = wmma16x16x32(a.v, b, acc);
    }
    int col = nt2 * 16 + lane15;
    if (col < LBL) {
      float bias2 = b2[col];
#pragma unroll
      for (int i = 0; i < 8; ++i) {
        int span = block0 + mt * 16 + laneh * 8 + i;
        if (span < N) out[(long)span * LBL + col] = acc[i] + bias2;
      }
    }
  }
}

extern "C" void kernel_launch(void* const* d_in, const int* in_sizes, int n_in,
                              void* d_out, int out_size, void* d_ws, size_t ws_size,
                              hipStream_t stream) {
  const float* xs        = (const float*)d_in[0];
  const int*   spans     = (const int*)d_in[1];
  const int*   batch_ids = (const int*)d_in[2];
  const float* W1        = (const float*)d_in[3];
  const float* b1        = (const float*)d_in[4];
  const float* W2        = (const float*)d_in[5];
  const float* b2        = (const float*)d_in[6];
  float* out = (float*)d_out;

  const int N = in_sizes[2];   // batch_ids count == number of spans

  _Float16* W1h = (_Float16*)d_ws;                      // 20480*16 halves
  _Float16* W2h = W1h + (size_t)W1H_ENTRIES * 16;       // 320*16 halves

  int prepThreads = 256;
  int prepBlocks  = (W1H_ENTRIES + W2H_ENTRIES + prepThreads - 1) / prepThreads;
  prep_weights<<<prepBlocks, prepThreads, 0, stream>>>(W1, W2, W1h, W2h);

  int blocks = (N + MT - 1) / MT;   // 2048 for N=65536
  span_mlp<<<blocks, 320, 0, stream>>>(xs, spans, batch_ids, W1h, b1, W2h, b2, out, N);
}
